// mmvaeplus_79207786873533
// MI455X (gfx1250) — matrix operand order repaired and checked by reading, not compiled, not run on hardware
//
#include <hip/hip_runtime.h>
#include <math.h>

typedef __attribute__((ext_vector_type(16))) _Float16 v16h;
typedef __attribute__((ext_vector_type(8)))  float    v8f;
typedef __attribute__((ext_vector_type(4)))  int      v4i;

struct __align__(16) U4 { unsigned a, b, c, d; };
struct __align__(16) F4 { float x, y, z, w; };
struct __align__(8)  H4 { _Float16 h0, h1, h2, h3; };
union AFrag { v16h v; U4 q[2]; };

#define EPSV 1e-08f

// ---- CDNA5 async global->LDS copy (ASYNCcnt path), with sync fallback ----
typedef __attribute__((address_space(1))) v4i as1_v4i;
typedef __attribute__((address_space(3))) v4i as3_v4i;

#if __has_builtin(__builtin_amdgcn_global_load_async_to_lds_b128)
#define HAS_ASYNC_LDS 1
__device__ __forceinline__ void async_cp16(const void* g, void* l) {
    __builtin_amdgcn_global_load_async_to_lds_b128((as1_v4i*)g, (as3_v4i*)l, 0, 0);
}
__device__ __forceinline__ void wait_async() {
#if __has_builtin(__builtin_amdgcn_s_wait_asynccnt)
    __builtin_amdgcn_s_wait_asynccnt(0);
#else
    asm volatile("s_wait_asynccnt 0x0" ::: "memory");
#endif
}
#else
#define HAS_ASYNC_LDS 0
__device__ __forceinline__ void async_cp16(const void* g, void* l) {
    *(U4*)l = *(const U4*)g;
}
__device__ __forceinline__ void wait_async() {}
#endif

__device__ __forceinline__ float softplusf(float v) {
    return fmaxf(v, 0.f) + log1pf(expf(-fabsf(v)));
}
__device__ __forceinline__ unsigned f2key(float f) {
    int b = __float_as_int(f);
    return (b >= 0) ? ((unsigned)b | 0x80000000u) : ~(unsigned)b;
}
__device__ __forceinline__ float key2f(unsigned k) {
    int b = (k & 0x80000000u) ? (int)(k & 0x7fffffffu) : (int)~k;
    return __int_as_float(b);
}

// ---------------- utility kernels ----------------
__global__ void fill_f32(float* p, float v, int n) {
    int i = blockIdx.x * blockDim.x + threadIdx.x;
    if (i < n) p[i] = v;
}
__global__ void fill_u32(unsigned* p, unsigned v, int n) {
    int i = blockIdx.x * blockDim.x + threadIdx.x;
    if (i < n) p[i] = v;
}
__global__ void init_bias(float* out, const float* bias, int rows, int C) {
    int i = blockIdx.x * blockDim.x + threadIdx.x;
    if (i < rows * C) out[i] = bias[i % C];
}
// pack f32 B (K x scols, row-major) into f16 Bt[n][k] (transposed), K zero-padded
// to Kpad, dst rows [row0, row0+nrows), source column = n (zero if n >= scols).
__global__ void pack_f16t(const float* __restrict__ src, int K, int scols,
                          _Float16* __restrict__ dst, int Kpad, int row0, int nrows) {
    int i = blockIdx.x * blockDim.x + threadIdx.x;
    if (i >= nrows * Kpad) return;
    int n = i / Kpad, k = i % Kpad;
    float v = (k < K && n < scols) ? src[(size_t)k * scols + n] : 0.f;
    dst[(size_t)(row0 + n) * Kpad + k] = (_Float16)v;
}
__global__ void gene_pre(const float* __restrict__ theta, float* lte, float* lgt, int G) {
    int g = blockIdx.x * blockDim.x + threadIdx.x;
    if (g >= G) return;
    float t = theta[g];
    lte[g] = logf(t + EPSV);
    lgt[g] = lgammaf(t);
}
__global__ void finalize_k(const float* scal, float* out, float invN) {
    if (threadIdx.x == 0 && blockIdx.x == 0)
        out[0] = (-scal[0] + scal[1]) * invN;
}

// ---------------- WMMA GEMM (double-buffered, async B staging) ----------------
// C[M x ldc] = A(f32, M x lda; f16 on the fly) @ B, Bt[n][k] pre-transposed f16.
// block = 64 threads (2 waves); wave w -> rows [by*32 + w*16), cols [bx*128, +128)
// REQUIRES: gridDim.y*32 == M, lda % 4 == 0, KP % 32 == 0.
__global__ __launch_bounds__(64)
void wmma_gemm_a32(const float* __restrict__ A, int lda,
                   const _Float16* __restrict__ Bt, int KP,
                   float* __restrict__ C, int ldc,
                   int Nvalid, int K,
                   const float* __restrict__ bias, int relu_a) {
    __shared__ __align__(32) _Float16 sA[2][32 * 32];
    __shared__ __align__(32) _Float16 sB[2][128 * 32]; // [n][k]
    const int tid = threadIdx.x, lane = tid & 31, wave = tid >> 5;
    const int rowBlk = blockIdx.y * 32;
    const int n0 = blockIdx.x * 128;

    auto stageA = [&](int buf, int k0) {
        if (k0 + 32 <= K) { // full chunk: batched b128 loads, then converts/stores
            F4 va[4];
#pragma unroll
            for (int i = 0; i < 4; ++i) {
                int idx = tid + i * 64;
                int r = idx >> 3, c = (idx & 7) << 2;
                va[i] = *(const F4*)(A + (size_t)(rowBlk + r) * lda + (k0 + c));
            }
#pragma unroll
            for (int i = 0; i < 4; ++i) {
                int idx = tid + i * 64;
                int r = idx >> 3, c = (idx & 7) << 2;
                F4 v = va[i];
                if (relu_a) {
                    v.x = fmaxf(v.x, 0.f); v.y = fmaxf(v.y, 0.f);
                    v.z = fmaxf(v.z, 0.f); v.w = fmaxf(v.w, 0.f);
                }
                H4 hv = {(_Float16)v.x, (_Float16)v.y, (_Float16)v.z, (_Float16)v.w};
                *(H4*)(&sA[buf][r * 32 + c]) = hv;
            }
        } else { // last partial chunk only
            for (int idx = tid; idx < 32 * 32; idx += 64) {
                int r = idx >> 5, k = idx & 31;
                float v = (k0 + k < K) ? A[(size_t)(rowBlk + r) * lda + (k0 + k)] : 0.f;
                if (relu_a) v = fmaxf(v, 0.f);
                sA[buf][r * 32 + k] = (_Float16)v;
            }
        }
    };
    auto stageB = [&](int buf, int k0) {
#pragma unroll
        for (int i = 0; i < 8; ++i) {
            int idx = tid + i * 64;
            int n = idx >> 2, j = (idx & 3) << 3; // halves
            async_cp16(Bt + (size_t)(n0 + n) * KP + (k0 + j), &sB[buf][n * 32 + j]);
        }
    };

    v8f acc[8] = {};
    const int nk = (K + 31) >> 5;
    stageB(0, 0);
    stageA(0, 0);
    wait_async();
    __syncthreads();
    for (int it = 0; it < nk; ++it) {
        const int buf = it & 1;
        if (it + 1 < nk) { // prefetch next chunk into other buffer
            stageB(buf ^ 1, (it + 1) * 32);
            stageA(buf ^ 1, (it + 1) * 32);
        }
        int r = lane & 15, hi = lane >> 4;
        AFrag af;
        af.q[0] = *(const U4*)(&sA[buf][(wave * 16 + r) * 32 + hi * 16]);
        af.q[1] = *(const U4*)(&sA[buf][(wave * 16 + r) * 32 + hi * 16 + 8]);
#pragma unroll
        for (int ct = 0; ct < 8; ++ct) {
            v16h bf = *(const v16h*)(&sB[buf][(ct * 16 + r) * 32 + hi * 16]);
            acc[ct] = __builtin_amdgcn_wmma_f32_16x16x32_f16(
                false, af.v, false, bf, (short)0, acc[ct], false, false);
        }
        wait_async();
        __syncthreads();
    }
    int r = lane & 15, rsel = (lane >> 4) * 8;
    int row0 = rowBlk + wave * 16;
#pragma unroll
    for (int ct = 0; ct < 8; ++ct) {
        int n = n0 + ct * 16 + r;
        if (n < Nvalid) {
            float bv = bias ? bias[n] : 0.f;
#pragma unroll
            for (int i = 0; i < 8; ++i)
                C[(size_t)(row0 + rsel + i) * ldc + n] = acc[ct][i] + bv;
        }
    }
}

// ---------------- fused mean/pi GEMMs + ZINB reduction ----------------
// Same tiling; two pre-transposed B matrices, epilogue evaluates log-ZINB
// against x and accumulates a scalar. REQUIRES gridDim.y*32 == M, K % 32 == 0.
__global__ __launch_bounds__(64)
void wmma_zinb(const float* __restrict__ A, int lda,            // hd (M x K)
               const _Float16* __restrict__ Bmt,
               const _Float16* __restrict__ Bpt, int KP,
               const float* __restrict__ x, int G,
               const float* __restrict__ bd2, const float* __restrict__ bdo,
               const float* __restrict__ theta,
               const float* __restrict__ lte, const float* __restrict__ lgt,
               int K, float* recon_acc) {
    __shared__ __align__(32) _Float16 sA[2][32 * 32];
    __shared__ __align__(32) _Float16 sBm[2][128 * 32];
    __shared__ __align__(32) _Float16 sBp[2][128 * 32];
    const int tid = threadIdx.x, lane = tid & 31, wave = tid >> 5;
    const int rowBlk = blockIdx.y * 32;
    const int n0 = blockIdx.x * 128;

    auto stageA = [&](int buf, int k0) {
        F4 va[4];
#pragma unroll
        for (int i = 0; i < 4; ++i) {
            int idx = tid + i * 64;
            int r = idx >> 3, c = (idx & 7) << 2;
            va[i] = *(const F4*)(A + (size_t)(rowBlk + r) * lda + (k0 + c));
        }
#pragma unroll
        for (int i = 0; i < 4; ++i) {
            int idx = tid + i * 64;
            int r = idx >> 3, c = (idx & 7) << 2;
            H4 hv = {(_Float16)va[i].x, (_Float16)va[i].y,
                     (_Float16)va[i].z, (_Float16)va[i].w};
            *(H4*)(&sA[buf][r * 32 + c]) = hv;
        }
    };
    auto stageB = [&](int buf, int k0) {
#pragma unroll
        for (int i = 0; i < 8; ++i) {
            int idx = tid + i * 64;
            int n = idx >> 2, j = (idx & 3) << 3;
            size_t go = (size_t)(n0 + n) * KP + (k0 + j);
            async_cp16(Bmt + go, &sBm[buf][n * 32 + j]);
            async_cp16(Bpt + go, &sBp[buf][n * 32 + j]);
        }
    };

    v8f am[8] = {};
    v8f ap[8] = {};
    const int nk = K >> 5;
    stageB(0, 0);
    stageA(0, 0);
    wait_async();
    __syncthreads();
    for (int it = 0; it < nk; ++it) {
        const int buf = it & 1;
        if (it + 1 < nk) {
            stageB(buf ^ 1, (it + 1) * 32);
            stageA(buf ^ 1, (it + 1) * 32);
        }
        int r = lane & 15, hi = lane >> 4;
        AFrag af;
        af.q[0] = *(const U4*)(&sA[buf][(wave * 16 + r) * 32 + hi * 16]);
        af.q[1] = *(const U4*)(&sA[buf][(wave * 16 + r) * 32 + hi * 16 + 8]);
#pragma unroll
        for (int ct = 0; ct < 8; ++ct) {
            v16h bm = *(const v16h*)(&sBm[buf][(ct * 16 + r) * 32 + hi * 16]);
            v16h bp = *(const v16h*)(&sBp[buf][(ct * 16 + r) * 32 + hi * 16]);
            am[ct] = __builtin_amdgcn_wmma_f32_16x16x32_f16(
                false, af.v, false, bm, (short)0, am[ct], false, false);
            ap[ct] = __builtin_amdgcn_wmma_f32_16x16x32_f16(
                false, af.v, false, bp, (short)0, ap[ct], false, false);
        }
        wait_async();
        __syncthreads();
    }
    // ZINB epilogue
    int r = lane & 15, rsel = (lane >> 4) * 8;
    int row0 = rowBlk + wave * 16;
    float lsum = 0.f;
#pragma unroll
    for (int ct = 0; ct < 8; ++ct) {
        int g = n0 + ct * 16 + r;
        if (g < G) {
            float b2v = bd2[g], bov = bdo[g], th = theta[g];
            float ltev = lte[g], lgtv = lgt[g];
#pragma unroll
            for (int i = 0; i < 8; ++i) {
                int row = row0 + rsel + i;
                float mu = softplusf(am[ct][i] + b2v);
                float pi = ap[ct][i] + bov;
                float xv = x[(size_t)row * G + g];
                float ltme = logf(th + mu + EPSV);
                float ptl = -pi + th * (ltev - ltme);
                float sppi = softplusf(-pi);
                float val;
                if (xv < EPSV)
                    val = softplusf(ptl) - sppi;
                else
                    val = -sppi + ptl + xv * (logf(mu + EPSV) - ltme)
                          + lgammaf(xv + th) - lgtv - lgammaf(xv + 1.f);
                lsum += val;
            }
        }
    }
    for (int off = 16; off; off >>= 1) lsum += __shfl_xor(lsum, off);
    if (lane == 0) atomicAdd(recon_acc, lsum);
}

// ---------------- GATv2 edge kernels ----------------
__global__ void edge_logits(const float* __restrict__ xl, const float* __restrict__ xr,
                            int ld, const float* __restrict__ att,
                            const int* __restrict__ src, const int* __restrict__ dst,
                            int E, int C, float* __restrict__ logit) {
    int w = (blockIdx.x * blockDim.x + threadIdx.x) >> 5;
    int lane = threadIdx.x & 31;
    if (w >= E) return;
    const float* pl = xl + (size_t)src[w] * ld;
    const float* pr = xr + (size_t)dst[w] * ld;
    float s = 0.f;
    for (int c = lane; c < C; c += 32) {
        float e = pl[c] + pr[c];
        e = e > 0.f ? e : 0.2f * e;
        s += e * att[c];
    }
    for (int off = 16; off; off >>= 1) s += __shfl_xor(s, off);
    if (lane == 0) logit[w] = s;
}
__global__ void seg_amax(const float* __restrict__ logit, const int* __restrict__ dst,
                         unsigned* __restrict__ segmax, int E) {
    int e = blockIdx.x * blockDim.x + threadIdx.x;
    if (e >= E) return;
    atomicMax(&segmax[dst[e]], f2key(logit[e]));
}
__global__ void edge_expsum(const float* __restrict__ logit, const int* __restrict__ dst,
                            const unsigned* __restrict__ segmax, float* __restrict__ segsum,
                            float* __restrict__ expo, int E) {
    int e = blockIdx.x * blockDim.x + threadIdx.x;
    if (e >= E) return;
    int d = dst[e];
    float ex = expf(logit[e] - key2f(segmax[d]));
    expo[e] = ex;
    atomicAdd(&segsum[d], ex);
}
__global__ void edge_scatter(const float* __restrict__ expo, const float* __restrict__ segsum,
                             const int* __restrict__ src, const int* __restrict__ dst,
                             const float* __restrict__ xl, int ld,
                             float* __restrict__ out, int C, int E) {
    int w = (blockIdx.x * blockDim.x + threadIdx.x) >> 5;
    int lane = threadIdx.x & 31;
    if (w >= E) return;
    int s = src[w], d = dst[w];
    float alpha = expo[w] / (segsum[d] + 1e-16f);
    const float* pl = xl + (size_t)s * ld;
    float* po = out + (size_t)d * C;
    for (int c = lane; c < C; c += 32) atomicAdd(&po[c], alpha * pl[c]);
}

// ---------------- reparameterize + KL ----------------
__global__ void z_kl_kernel(const float* __restrict__ h2, const float* __restrict__ noise,
                            float* __restrict__ z, float* kl_acc, int Nrows) {
    int idx = blockIdx.x * blockDim.x + threadIdx.x;
    float kl = 0.f;
    if (idx < Nrows * 32) {
        int row = idx >> 5, j = idx & 31;
        float mu = h2[row * 64 + j];
        float lv = h2[row * 64 + 32 + j];
        z[idx] = mu + noise[idx] * expf(0.5f * lv);
        kl = 0.5f * (mu * mu + expf(lv) - lv - 1.f);
    }
    for (int off = 16; off; off >>= 1) kl += __shfl_xor(kl, off);
    if ((threadIdx.x & 31) == 0) atomicAdd(kl_acc, kl);
}

extern "C" void kernel_launch(void* const* d_in, const int* in_sizes, int n_in,
                              void* d_out, int out_size, void* d_ws, size_t ws_size,
                              hipStream_t stream) {
    const int N = 20000, G = 3000, HID = 256, LAT = 32;
    const int KP1 = 3008;  // K of layer-1 GEMM padded to 32
    const int GP = 3072;   // gene dim padded to 128
    const float* x    = (const float*)d_in[0];
    const int*   ei   = (const int*)d_in[1];
    const float* noise= (const float*)d_in[2];
    const float* Wl1  = (const float*)d_in[3];
    const float* Wr1  = (const float*)d_in[4];
    const float* att1 = (const float*)d_in[5];
    const float* b1   = (const float*)d_in[6];
    const float* Wl2  = (const float*)d_in[7];
    const float* Wr2  = (const float*)d_in[8];
    const float* att2 = (const float*)d_in[9];
    const float* b2   = (const float*)d_in[10];
    const float* Wd1  = (const float*)d_in[11];
    const float* bd1  = (const float*)d_in[12];
    const float* Wd2  = (const float*)d_in[13];
    const float* bd2  = (const float*)d_in[14];
    const float* Wdo  = (const float*)d_in[15];
    const float* bdo  = (const float*)d_in[16];
    const float* theta= (const float*)d_in[17];
    const int E = in_sizes[1] / 2;
    const int* srcI = ei;
    const int* dstI = ei + E;

    char* ws = (char*)d_ws;
    size_t off = 0;
    auto alloc = [&](size_t bytes) { size_t o = off; off = (off + bytes + 255) & ~(size_t)255; return o; };
    size_t o_w1t   = alloc((size_t)512 * KP1 * 2);   // [n=512][k=3008]
    size_t o_xlr1  = alloc((size_t)N * 512 * 4);
    size_t o_h     = alloc((size_t)N * HID * 4);
    size_t o_w2t   = alloc((size_t)128 * HID * 2);   // [n=128][k=256]
    size_t o_xlr2  = alloc((size_t)N * 128 * 4);
    size_t o_h2    = alloc((size_t)N * 64 * 4);
    size_t o_z     = alloc((size_t)N * LAT * 4);
    size_t o_wd1t  = alloc((size_t)HID * LAT * 2);   // [n=256][k=32]
    size_t o_hd    = alloc((size_t)N * HID * 4);
    size_t o_wd2t  = alloc((size_t)GP * HID * 2);    // [n=3072][k=256]
    size_t o_wdot  = alloc((size_t)GP * HID * 2);
    size_t o_lte   = alloc((size_t)G * 4);
    size_t o_lgt   = alloc((size_t)G * 4);
    size_t o_logit = alloc((size_t)E * 4);
    size_t o_expo  = alloc((size_t)E * 4);
    size_t o_segmx = alloc((size_t)N * 4);
    size_t o_segsm = alloc((size_t)N * 4);
    size_t o_scal  = alloc(8);
    (void)ws_size;

    _Float16* w1t  = (_Float16*)(ws + o_w1t);
    float*    xlr1 = (float*)(ws + o_xlr1);
    float*    h    = (float*)(ws + o_h);
    _Float16* w2t  = (_Float16*)(ws + o_w2t);
    float*    xlr2 = (float*)(ws + o_xlr2);
    float*    h2   = (float*)(ws + o_h2);
    float*    z    = (float*)(ws + o_z);
    _Float16* wd1t = (_Float16*)(ws + o_wd1t);
    float*    hd   = (float*)(ws + o_hd);
    _Float16* wd2t = (_Float16*)(ws + o_wd2t);
    _Float16* wdot = (_Float16*)(ws + o_wdot);
    float*    lte  = (float*)(ws + o_lte);
    float*    lgt  = (float*)(ws + o_lgt);
    float*    logit= (float*)(ws + o_logit);
    float*    expo = (float*)(ws + o_expo);
    unsigned* segmx= (unsigned*)(ws + o_segmx);
    float*    segsm= (float*)(ws + o_segsm);
    float*    scal = (float*)(ws + o_scal); // [0]=recon sum, [1]=kl sum

    const int TB = 256;
    auto cdiv = [](long long a, long long b) { return (unsigned)((a + b - 1) / b); };
    const unsigned eblk  = cdiv(E, TB);
    const unsigned ewblk = cdiv((long long)E * 32, TB);

    // ---- layer 1: xlr1 = x @ [Wl1|Wr1] ----
    pack_f16t<<<cdiv((long long)256 * KP1, TB), TB, 0, stream>>>(Wl1, G, HID, w1t, KP1, 0, 256);
    pack_f16t<<<cdiv((long long)256 * KP1, TB), TB, 0, stream>>>(Wr1, G, HID, w1t, KP1, 256, 256);
    wmma_gemm_a32<<<dim3(4, 625), 64, 0, stream>>>(x, G, w1t, KP1, xlr1, 512,
                                                   512, G, nullptr, 0);
    fill_u32<<<cdiv(N, TB), TB, 0, stream>>>(segmx, 0x007FFFFFu, N); // key(-inf)
    fill_f32<<<cdiv(N, TB), TB, 0, stream>>>(segsm, 0.f, N);
    edge_logits<<<ewblk, TB, 0, stream>>>(xlr1, xlr1 + 256, 512, att1, srcI, dstI, E, HID, logit);
    seg_amax<<<eblk, TB, 0, stream>>>(logit, dstI, segmx, E);
    edge_expsum<<<eblk, TB, 0, stream>>>(logit, dstI, segmx, segsm, expo, E);
    init_bias<<<cdiv((long long)N * HID, TB), TB, 0, stream>>>(h, b1, N, HID);
    edge_scatter<<<ewblk, TB, 0, stream>>>(expo, segsm, srcI, dstI, xlr1, 512, h, HID, E);

    // ---- layer 2: xlr2 = relu(h) @ [Wl2|Wr2] ----
    pack_f16t<<<cdiv((long long)64 * HID, TB), TB, 0, stream>>>(Wl2, HID, 64, w2t, HID, 0, 64);
    pack_f16t<<<cdiv((long long)64 * HID, TB), TB, 0, stream>>>(Wr2, HID, 64, w2t, HID, 64, 64);
    wmma_gemm_a32<<<dim3(1, 625), 64, 0, stream>>>(h, HID, w2t, HID, xlr2, 128,
                                                   128, HID, nullptr, 1);
    fill_u32<<<cdiv(N, TB), TB, 0, stream>>>(segmx, 0x007FFFFFu, N);
    fill_f32<<<cdiv(N, TB), TB, 0, stream>>>(segsm, 0.f, N);
    edge_logits<<<ewblk, TB, 0, stream>>>(xlr2, xlr2 + 64, 128, att2, srcI, dstI, E, 64, logit);
    seg_amax<<<eblk, TB, 0, stream>>>(logit, dstI, segmx, E);
    edge_expsum<<<eblk, TB, 0, stream>>>(logit, dstI, segmx, segsm, expo, E);
    init_bias<<<cdiv((long long)N * 64, TB), TB, 0, stream>>>(h2, b2, N, 64);
    edge_scatter<<<ewblk, TB, 0, stream>>>(expo, segsm, srcI, dstI, xlr2, 128, h2, 64, E);

    // ---- reparameterize + KL ----
    fill_f32<<<1, 32, 0, stream>>>(scal, 0.f, 2);
    z_kl_kernel<<<cdiv((long long)N * 32, TB), TB, 0, stream>>>(h2, noise, z, scal + 1, N);

    // ---- decoder: hd = z @ Wd1 + bd1 ----
    pack_f16t<<<cdiv((long long)HID * LAT, TB), TB, 0, stream>>>(Wd1, LAT, HID, wd1t, LAT, 0, HID);
    wmma_gemm_a32<<<dim3(2, 625), 64, 0, stream>>>(z, LAT, wd1t, LAT, hd, HID,
                                                   HID, LAT, bd1, 0);

    // ---- fused mean/pi GEMMs + ZINB reduction ----
    pack_f16t<<<cdiv((long long)GP * HID, TB), TB, 0, stream>>>(Wd2, HID, G, wd2t, HID, 0, GP);
    pack_f16t<<<cdiv((long long)GP * HID, TB), TB, 0, stream>>>(Wdo, HID, G, wdot, HID, 0, GP);
    gene_pre<<<cdiv(G, TB), TB, 0, stream>>>(theta, lte, lgt, G);
    wmma_zinb<<<dim3(GP / 128, 625), 64, 0, stream>>>(hd, HID, wd2t, wdot, HID,
                                                      x, G, bd2, bdo, theta, lte, lgt,
                                                      HID, scal);

    finalize_k<<<1, 32, 0, stream>>>(scal, (float*)d_out, 1.f / (float)N);
}
